// Transformer_50843822850105
// MI455X (gfx1250) — compile-verified, matrix operand order checked by reading, not run on hardware
//
#include <hip/hip_runtime.h>
#include <hip/hip_bf16.h>

// ---------------------------------------------------------------------------
// Transformer forward (live subgraph only) for MI455X / gfx1250.
//
// Live dataflow (reference has dead encoder + only last decoder layer lives):
//   tgt -> embed*(1+sqrt(D)) + PE -> dec[3]{self-attn(causal), cross-attn,
//   FFN} -> dec_ln -> logits (2048x512x32000) -> softmax -> d_out
//
// All GEMMs use v_wmma_f32_16x16x32_f16 (f32 accumulate). Weights/activations
// are cast to f16 in workspace; B operands are stored transposed [N][K] so
// both A and B WMMA fragments are two contiguous 16B loads per lane.
// ---------------------------------------------------------------------------

typedef __attribute__((ext_vector_type(16))) _Float16 v16h;
typedef __attribute__((ext_vector_type(8)))  _Float16 v8h;
typedef __attribute__((ext_vector_type(8)))  float    v8f;

#define B_  4
#define S_  512
#define D_  512
#define H_  8
#define DH_ 64
#define FF_ 2048
#define V_  32000
#define NT_ (B_ * S_)   // 2048 token rows

// ---------------- WMMA fragment load -----------------------------------------
// 16-bit A-matrix 16x32 lane layout (ISA 7.12.2): lanes 0-15 row m hold
// K=0..7 (VGPR0-3) and K=16..23 (VGPR4-7); lanes 16-31 hold K=8..15, K=24..31.
// With row-major [*,K] f16 data that is exactly two 8-half contiguous chunks.
__device__ __forceinline__ v16h ld_frag16(const _Float16* p) {
  v8h lo = *reinterpret_cast<const v8h*>(p);
  v8h hi = *reinterpret_cast<const v8h*>(p + 16);
  v16h r;
#pragma unroll
  for (int i = 0; i < 8; ++i) { r[i] = lo[i]; r[i + 8] = hi[i]; }
  return r;
}

// ---------------- Generic batched WMMA GEMM ----------------------------------
// C[z] = alpha * A[z] x B[z]^T + bias  (optional relu)
// A: f16 [M,K] row-major (lda), Bt: f16 [N,K] row-major (ldb = K stride),
// C: f32, per-z offset = (z/divC)*sC1 + (z%divC)*sC2 (lets attention heads
// scatter into the concat layout [B,S,H*DH]).
// Block = 256 threads = 8 waves (2 wave-rows x 4 wave-cols) -> 32x256 tile,
// each wave computes 16x64 via 4 WMMA accumulators sharing one A fragment.
__global__ __launch_bounds__(256) void gemm_wmma(
    const _Float16* __restrict__ A, long long sAz, int lda,
    const _Float16* __restrict__ Bt, long long sBz, int ldb,
    float* __restrict__ C, int divC, long long sC1, long long sC2, int ldc,
    const float* __restrict__ bias, int M, int N, int K, float alpha, int relu)
{
  int z = blockIdx.z;
  const _Float16* Az = A + (long long)z * sAz;
  const _Float16* Bz = Bt + (long long)z * sBz;
  float* Cz = C + (long long)(z / divC) * sC1 + (long long)(z % divC) * sC2;

  int wave = threadIdx.x >> 5;
  int lane = threadIdx.x & 31;
  int mBase = blockIdx.y * 32 + (wave >> 2) * 16;
  int nBase = blockIdx.x * 256 + (wave & 3) * 64;
  if (mBase >= M || nBase >= N) return;   // wave-uniform: EXEC stays all-ones

  int lrow = lane & 15;
  int koff = (lane >> 4) << 3;            // 0 or 8

  const _Float16* aP  = Az + (long long)(mBase + lrow) * lda + koff;
  const _Float16* bP0 = Bz + (long long)(nBase +  0 + lrow) * ldb + koff;
  const _Float16* bP1 = Bz + (long long)(nBase + 16 + lrow) * ldb + koff;
  const _Float16* bP2 = Bz + (long long)(nBase + 32 + lrow) * ldb + koff;
  const _Float16* bP3 = Bz + (long long)(nBase + 48 + lrow) * ldb + koff;

  v8f acc0 = {0.f,0.f,0.f,0.f,0.f,0.f,0.f,0.f};
  v8f acc1 = acc0, acc2 = acc0, acc3 = acc0;

  for (int k0 = 0; k0 < K; k0 += 32) {
    v16h a  = ld_frag16(aP  + k0);
    v16h b0 = ld_frag16(bP0 + k0);
    v16h b1 = ld_frag16(bP1 + k0);
    v16h b2 = ld_frag16(bP2 + k0);
    v16h b3 = ld_frag16(bP3 + k0);
    acc0 = __builtin_amdgcn_wmma_f32_16x16x32_f16(false, a, false, b0, (short)0, acc0, false, false);
    acc1 = __builtin_amdgcn_wmma_f32_16x16x32_f16(false, a, false, b1, (short)0, acc1, false, false);
    acc2 = __builtin_amdgcn_wmma_f32_16x16x32_f16(false, a, false, b2, (short)0, acc2, false, false);
    acc3 = __builtin_amdgcn_wmma_f32_16x16x32_f16(false, a, false, b3, (short)0, acc3, false, false);
  }

  // f32 C/D layout: VGPR r: lanes 0-15 -> M=r, lanes 16-31 -> M=8+r; N=lane&15
  int rowAdd = (lane >> 4) * 8;
#pragma unroll
  for (int r = 0; r < 8; ++r) {
    long long rowB = (long long)(mBase + r + rowAdd) * ldc;
    int c0 = nBase + lrow;
    float v0 = alpha * acc0[r] + (bias ? bias[c0     ] : 0.0f);
    float v1 = alpha * acc1[r] + (bias ? bias[c0 + 16] : 0.0f);
    float v2 = alpha * acc2[r] + (bias ? bias[c0 + 32] : 0.0f);
    float v3 = alpha * acc3[r] + (bias ? bias[c0 + 48] : 0.0f);
    if (relu) { v0=fmaxf(v0,0.f); v1=fmaxf(v1,0.f); v2=fmaxf(v2,0.f); v3=fmaxf(v3,0.f); }
    Cz[rowB + c0     ] = v0;
    Cz[rowB + c0 + 16] = v1;
    Cz[rowB + c0 + 32] = v2;
    Cz[rowB + c0 + 48] = v3;
  }
}

// ---------------- Elementwise / reshape / cast kernels -----------------------

// emb gather + positional encoding: te = e*(1+sqrt(D)) + PE[t,d]; writes f32+f16
__global__ __launch_bounds__(256) void embed_pe(
    const int* __restrict__ tok, const float* __restrict__ emb,
    float* __restrict__ te32, _Float16* __restrict__ te16)
{
  long long row = blockIdx.x;
  int t = (int)(row % S_);
  int token = tok[row];
  const float* e = emb + (long long)token * D_;
  long long base = row * D_;
  const float k1   = 23.627416997969522f;              // 1 + sqrt(512)
  const float nlog = -(2.0f / D_) * 9.210340371976184f; // -(2/D) ln 10000
  for (int d = threadIdx.x; d < D_; d += 256) {
    float ang = (float)t * expf(nlog * (float)d);
    float pe  = ((d & 1) == 0) ? sinf(ang) : cosf(ang);
    float v   = e[d] * k1 + pe;
    te32[base + d] = v;
    te16[base + d] = (_Float16)v;
  }
}

// per-head projection weight (H,D,DH) f32 -> transposed f16 [N=H*DH][D]
__global__ __launch_bounds__(256) void cast_headw(
    const float* __restrict__ w, _Float16* __restrict__ wt)
{
  int idx = blockIdx.x * 256 + threadIdx.x;   // n*D + d
  int n = idx >> 9, d = idx & (D_ - 1);
  wt[idx] = (_Float16)w[(n >> 6) * (D_ * DH_) + d * DH_ + (n & (DH_ - 1))];
}

// f32 [R,C] row-major -> f16 [C,R] row-major (GEMM B-operand form)
__global__ __launch_bounds__(256) void cast_transpose(
    const float* __restrict__ in, _Float16* __restrict__ out, int R, int C)
{
  long long idx = (long long)blockIdx.x * 256 + threadIdx.x;
  if (idx >= (long long)R * C) return;
  int c = (int)(idx / R), r = (int)(idx % R);
  out[idx] = (_Float16)in[(long long)r * C + c];
}

__global__ __launch_bounds__(256) void cast_f16(
    const float* __restrict__ in, _Float16* __restrict__ out, long long n)
{
  long long i = (long long)blockIdx.x * 256 + threadIdx.x;
  if (i < n) out[i] = (_Float16)in[i];
}

// [B,S,H*DH] f32 -> [B,H,S,DH] f16   (Q / K head split)
__global__ __launch_bounds__(256) void reshape_heads(
    const float* __restrict__ in, _Float16* __restrict__ out)
{
  long long o = (long long)blockIdx.x * 256 + threadIdx.x; // ((b*H+h)*S+s)*DH+k
  int k = (int)(o & (DH_ - 1));
  long long t = o >> 6;
  int s = (int)(t & (S_ - 1));
  long long bh = t >> 9;
  int h = (int)(bh & (H_ - 1));
  int b = (int)(bh >> 3);
  out[o] = (_Float16)in[(long long)(b * S_ + s) * D_ + h * DH_ + k];
}

// [B,S,H*DH] f32 -> [B,H,DH,S] f16   (V transposed: B-operand for W x V)
__global__ __launch_bounds__(256) void reshape_headsT(
    const float* __restrict__ in, _Float16* __restrict__ out)
{
  long long o = (long long)blockIdx.x * 256 + threadIdx.x; // ((b*H+h)*DH+k)*S+s
  int s = (int)(o & (S_ - 1));
  long long t = o >> 9;
  int k = (int)(t & (DH_ - 1));
  long long bh = t >> 6;
  int h = (int)(bh & (H_ - 1));
  int b = (int)(bh >> 3);
  out[o] = (_Float16)in[(long long)(b * S_ + s) * D_ + h * DH_ + k];
}

// LayerNorm over D=512 with optional residual add; writes f32 + f16 copies.
__global__ __launch_bounds__(256) void layernorm512(
    const float* __restrict__ x, const float* __restrict__ resid,
    const float* __restrict__ g, const float* __restrict__ b,
    float* __restrict__ y32, _Float16* __restrict__ y16)
{
  __shared__ float red[256];
  long long base = (long long)blockIdx.x * D_;
  int tid = threadIdx.x;
  float v0 = x[base + tid], v1 = x[base + tid + 256];
  if (resid) { v0 += resid[base + tid]; v1 += resid[base + tid + 256]; }
  red[tid] = v0 + v1; __syncthreads();
  for (int s = 128; s > 0; s >>= 1) { if (tid < s) red[tid] += red[tid + s]; __syncthreads(); }
  float mean = red[0] * (1.0f / D_); __syncthreads();
  float d0 = v0 - mean, d1 = v1 - mean;
  red[tid] = d0 * d0 + d1 * d1; __syncthreads();
  for (int s = 128; s > 0; s >>= 1) { if (tid < s) red[tid] += red[tid + s]; __syncthreads(); }
  float rstd = rsqrtf(red[0] * (1.0f / D_) + 1e-5f);
  float o0 = d0 * rstd * g[tid]       + b[tid];
  float o1 = d1 * rstd * g[tid + 256] + b[tid + 256];
  y32[base + tid]       = o0;  y32[base + tid + 256] = o1;
  y16[base + tid]       = (_Float16)o0;
  y16[base + tid + 256] = (_Float16)o1;
}

// Row softmax over 512 attention scores (optional causal mask), emits f16.
__global__ __launch_bounds__(256) void softmax_scores(
    const float* __restrict__ sc, _Float16* __restrict__ w, int causal)
{
  __shared__ float red[256];
  long long row = blockIdx.x;          // z*S + s
  int srow = (int)(row % S_);
  long long base = row * S_;
  int tid = threadIdx.x;
  float a0 = sc[base + tid], a1 = sc[base + tid + 256];
  if (causal) {
    if (tid > srow)       a0 = -1e20f;
    if (tid + 256 > srow) a1 = -1e20f;
  }
  red[tid] = fmaxf(a0, a1); __syncthreads();
  for (int s = 128; s > 0; s >>= 1) { if (tid < s) red[tid] = fmaxf(red[tid], red[tid + s]); __syncthreads(); }
  float m = red[0]; __syncthreads();
  float e0 = expf(a0 - m), e1 = expf(a1 - m);
  red[tid] = e0 + e1; __syncthreads();
  for (int s = 128; s > 0; s >>= 1) { if (tid < s) red[tid] += red[tid + s]; __syncthreads(); }
  float inv = 1.0f / red[0];
  w[base + tid]       = (_Float16)(e0 * inv);
  w[base + tid + 256] = (_Float16)(e1 * inv);
}

// Final vocab softmax, one block per token row, V=32000 columns.
__global__ __launch_bounds__(512) void softmax_vocab(
    const float* __restrict__ logits, float* __restrict__ out)
{
  __shared__ float red[512];
  long long row = blockIdx.x;
  const float* x = logits + row * (long long)V_;
  float* o = out + row * (long long)V_;
  int tid = threadIdx.x;
  float m = -3.4e38f;
  for (int j = tid; j < V_; j += 512) m = fmaxf(m, x[j]);
  red[tid] = m; __syncthreads();
  for (int s = 256; s > 0; s >>= 1) { if (tid < s) red[tid] = fmaxf(red[tid], red[tid + s]); __syncthreads(); }
  m = red[0]; __syncthreads();
  float sum = 0.0f;
  for (int j = tid; j < V_; j += 512) sum += expf(x[j] - m);
  red[tid] = sum; __syncthreads();
  for (int s = 256; s > 0; s >>= 1) { if (tid < s) red[tid] += red[tid + s]; __syncthreads(); }
  float inv = 1.0f / red[0];
  for (int j = tid; j < V_; j += 512) o[j] = expf(x[j] - m) * inv;
}

// ---------------- Host orchestration -----------------------------------------
// Param leaf indexing: d_in = [src, tgt, tgt_mask, <params leaves>] with
// params flattened in sorted-key order (jax tree flatten):
//   dec (4 x 26 leaves: ca{kb,kw,ob,ow,qb,qw,vb,vw}, ffn{b1,b2,w1,w2},
//        ln1{b,g}, ln2{b,g}, ln3{b,g}, sa{...}), dec_ln{b,g}, emb,
//   enc (64), enc_ln{b,g}, out_b, out_w.
enum {
  IN_TGT = 1, P0 = 3,
  DEC3  = P0 + 3 * 26,                       // last decoder layer = leaf 81
  CA_KB = DEC3, CA_KW, CA_OB, CA_OW, CA_QB, CA_QW, CA_VB, CA_VW,
  FF_B1, FF_B2, FF_W1, FF_W2,
  LN1_B, LN1_G, LN2_B, LN2_G, LN3_B, LN3_G,
  SA_KB, SA_KW, SA_OB, SA_OW, SA_QB, SA_QW, SA_VB, SA_VW,
  DLN_B = P0 + 104, DLN_G = P0 + 105, EMB_I = P0 + 106,
  OUT_B = P0 + 173, OUT_W = P0 + 174,
};

extern "C" void kernel_launch(void* const* d_in, const int* in_sizes, int n_in,
                              void* d_out, int out_size, void* d_ws, size_t ws_size,
                              hipStream_t stream) {
  (void)in_sizes; (void)n_in; (void)out_size; (void)ws_size;
  auto F = [&](int i) { return (const float*)d_in[i]; };
  const int* tgt = (const int*)d_in[IN_TGT];

  // ---- workspace carve-up (~440 MB; 256B aligned) ----
  char* Wb = (char*)d_ws;
  size_t off = 0;
  auto alloc = [&](size_t bytes) -> void* {
    void* p = Wb + off;
    off = (off + bytes + 255) & ~(size_t)255;
    return p;
  };
  const long long ND  = (long long)NT_ * D_;       // 1M elems
  const long long NSS = (long long)B_ * H_ * S_ * S_; // 8.4M elems
  float*    te32  = (float*)   alloc(ND * 4);
  _Float16* te16  = (_Float16*)alloc(ND * 2);
  _Float16* wq16  = (_Float16*)alloc((size_t)D_ * D_ * 2);
  _Float16* wk16  = (_Float16*)alloc((size_t)D_ * D_ * 2);
  _Float16* wv16  = (_Float16*)alloc((size_t)D_ * D_ * 2);
  _Float16* wo16  = (_Float16*)alloc((size_t)D_ * D_ * 2);
  float*    q32   = (float*)   alloc(ND * 4);
  float*    k32   = (float*)   alloc(ND * 4);
  float*    v32   = (float*)   alloc(ND * 4);
  _Float16* qh16  = (_Float16*)alloc(ND * 2);
  _Float16* kh16  = (_Float16*)alloc(ND * 2);
  _Float16* vt16  = (_Float16*)alloc(ND * 2);
  float*    sc32  = (float*)   alloc(NSS * 4);
  _Float16* wsm16 = (_Float16*)alloc(NSS * 2);
  float*    ao32  = (float*)   alloc(ND * 4);
  _Float16* ao16  = (_Float16*)alloc(ND * 2);
  float*    x32   = (float*)   alloc(ND * 4);
  float*    n1_32 = (float*)   alloc(ND * 4);
  _Float16* n1_16 = (_Float16*)alloc(ND * 2);
  float*    n2_32 = (float*)   alloc(ND * 4);
  _Float16* n2_16 = (_Float16*)alloc(ND * 2);
  _Float16* w1T   = (_Float16*)alloc((size_t)D_ * FF_ * 2);
  _Float16* w2T   = (_Float16*)alloc((size_t)D_ * FF_ * 2);
  float*    h32   = (float*)   alloc((long long)NT_ * FF_ * 4);
  _Float16* h16   = (_Float16*)alloc((long long)NT_ * FF_ * 2);
  float*    f32o  = (float*)   alloc(ND * 4);
  float*    o32   = (float*)   alloc(ND * 4);
  _Float16* o16   = (_Float16*)alloc(ND * 2);
  float*    d32   = (float*)   alloc(ND * 4);
  _Float16* d16   = (_Float16*)alloc(ND * 2);
  _Float16* owT   = (_Float16*)alloc((size_t)V_ * D_ * 2);
  float*    logit = (float*)   alloc((long long)NT_ * V_ * 4);

  auto gemm = [&](const _Float16* A, long long sAz, int lda,
                  const _Float16* Bt, long long sBz, int ldb,
                  float* C, int divC, long long sC1, long long sC2, int ldc,
                  const float* bias, int M, int N, int K, float alpha, int relu, int Z) {
    dim3 g((N + 255) / 256, (M + 31) / 32, Z);
    gemm_wmma<<<g, 256, 0, stream>>>(A, sAz, lda, Bt, sBz, ldb,
                                     C, divC, sC1, sC2, ldc, bias, M, N, K, alpha, relu);
  };

  const float INV_SQRT_D = 0.044194173824159216f;  // 1/sqrt(512)

  // ---- embedding + positional encoding ----
  embed_pe<<<NT_, 256, 0, stream>>>(tgt, F(EMB_I), te32, te16);

  // ---- one full MHA block (shared by self- and cross-attention) ----
  auto attention = [&](const _Float16* xin16, const float* resid,
                       int QW, int QB, int KW, int KB, int VW, int VB, int OW, int OB,
                       int LNG, int LNB, int causal,
                       float* out32, _Float16* out16) {
    cast_headw   <<<(D_ * D_) / 256, 256, 0, stream>>>(F(QW), wq16);
    cast_headw   <<<(D_ * D_) / 256, 256, 0, stream>>>(F(KW), wk16);
    cast_headw   <<<(D_ * D_) / 256, 256, 0, stream>>>(F(VW), wv16);
    cast_transpose<<<(D_ * D_) / 256, 256, 0, stream>>>(F(OW), wo16, D_, D_);
    // fused-head projections: [2048,512] x [512,512]
    gemm(xin16, 0, D_, wq16, 0, D_, q32, 1, 0, 0, D_, F(QB), NT_, D_, D_, 1.f, 0, 1);
    gemm(xin16, 0, D_, wk16, 0, D_, k32, 1, 0, 0, D_, F(KB), NT_, D_, D_, 1.f, 0, 1);
    gemm(xin16, 0, D_, wv16, 0, D_, v32, 1, 0, 0, D_, F(VB), NT_, D_, D_, 1.f, 0, 1);
    reshape_heads <<<(int)(ND / 256), 256, 0, stream>>>(q32, qh16);
    reshape_heads <<<(int)(ND / 256), 256, 0, stream>>>(k32, kh16);
    reshape_headsT<<<(int)(ND / 256), 256, 0, stream>>>(v32, vt16);
    // scores: per (b,h): [512,64] x [64,512] scaled by 1/sqrt(D)
    gemm(qh16, (long long)S_ * DH_, DH_, kh16, (long long)S_ * DH_, DH_,
         sc32, 1, (long long)S_ * S_, 0, S_, nullptr, S_, S_, DH_, INV_SQRT_D, 0, B_ * H_);
    softmax_scores<<<B_ * H_ * S_, 256, 0, stream>>>(sc32, wsm16, causal);
    // O = W x V, scattered to head-concat layout [B,S,H*DH]
    gemm(wsm16, (long long)S_ * S_, S_, vt16, (long long)DH_ * S_, S_,
         ao32, H_, (long long)S_ * D_, DH_, D_, nullptr, S_, DH_, S_, 1.f, 0, B_ * H_);
    cast_f16<<<(int)(ND / 256), 256, 0, stream>>>(ao32, ao16, ND);
    // output projection + residual layernorm
    gemm(ao16, 0, D_, wo16, 0, D_, x32, 1, 0, 0, D_, F(OB), NT_, D_, D_, 1.f, 0, 1);
    layernorm512<<<NT_, 256, 0, stream>>>(x32, resid, F(LNG), F(LNB), out32, out16);
  };

  // ---- last decoder layer (only one that matters) ----
  attention(te16, te32, SA_QW, SA_QB, SA_KW, SA_KB, SA_VW, SA_VB, SA_OW, SA_OB,
            LN1_G, LN1_B, /*causal=*/1, n1_32, n1_16);
  attention(n1_16, n1_32, CA_QW, CA_QB, CA_KW, CA_KB, CA_VW, CA_VB, CA_OW, CA_OB,
            LN2_G, LN2_B, /*causal=*/0, n2_32, n2_16);

  // ---- FFN: relu(x W1 + b1) W2 + b2, then ln3(+resid), then dec_ln ----
  cast_transpose<<<(D_ * FF_) / 256, 256, 0, stream>>>(F(FF_W1), w1T, D_, FF_);
  gemm(n2_16, 0, D_, w1T, 0, D_, h32, 1, 0, 0, FF_, F(FF_B1), NT_, FF_, D_, 1.f, 1, 1);
  cast_f16<<<(int)(((long long)NT_ * FF_) / 256), 256, 0, stream>>>(h32, h16, (long long)NT_ * FF_);
  cast_transpose<<<(FF_ * D_) / 256, 256, 0, stream>>>(F(FF_W2), w2T, FF_, D_);
  gemm(h16, 0, FF_, w2T, 0, FF_, f32o, 1, 0, 0, D_, F(FF_B2), NT_, D_, FF_, 1.f, 0, 1);
  layernorm512<<<NT_, 256, 0, stream>>>(f32o, n2_32, F(LN3_G), F(LN3_B), o32, o16);
  layernorm512<<<NT_, 256, 0, stream>>>(o32, nullptr, F(DLN_G), F(DLN_B), d32, d16);

  // ---- logits + vocab softmax (HBM-bound stage) ----
  cast_transpose<<<(int)(((long long)D_ * V_) / 256), 256, 0, stream>>>(F(OUT_W), owT, D_, V_);
  gemm(d16, 0, D_, owT, 0, D_, logit, 1, 0, 0, V_, F(OUT_B), NT_, V_, D_, 1.f, 0, 1);
  softmax_vocab<<<NT_, 512, 0, stream>>>(logit, (float*)d_out);
}
